// AngularPrototypicalLoss_61392262529414
// MI455X (gfx1250) — matrix operand order repaired and checked by reading, not compiled, not run on hardware
//
#include <hip/hip_runtime.h>
#include <math.h>

typedef __attribute__((ext_vector_type(16))) _Float16 v16h;
typedef __attribute__((ext_vector_type(8)))  _Float16 v8h;
typedef __attribute__((ext_vector_type(8)))  float    v8f;
typedef unsigned int v4u __attribute__((ext_vector_type(4)));
typedef int          v8i __attribute__((ext_vector_type(8)));
typedef int          v4i __attribute__((ext_vector_type(4)));

#define NROWS 65536
#define DIM   256
#define NCLS  1024
#define COS_M_  0.9800665778412416f
#define SIN_M_  0.19866933079506122f
#define TH_     (-0.9800665778412416f)   /* cos(pi - m) */
#define MARGIN_ 0.2f
#define INV_T_  10.0f

#define CHUNK_ROWS 32
#define NCHUNK (NCLS / CHUNK_ROWS)       /* 32 */
#define LROW 264                          /* padded LDS row stride in f16 (528B) */

#if __has_builtin(__builtin_amdgcn_tensor_load_to_lds)
#define HAVE_TDM 1
#else
#define HAVE_TDM 0
#endif

// ---------------- zero init (ws is poisoned / stale between calls) ----------
__global__ void zero_f32(float* __restrict__ p, int n) {
    int i = blockIdx.x * 256 + threadIdx.x;
    if (i < n) p[i] = 0.0f;
}

// ---- Kernel A: L2-normalize rows, emit f16 copy, atomic scatter into protos
__global__ __launch_bounds__(256)
void normalize_scatter(const float* __restrict__ emb,
                       const int*   __restrict__ labels,
                       _Float16*    __restrict__ eh,
                       float*       __restrict__ protoSum,
                       float*       __restrict__ counts)
{
    const int wid  = threadIdx.x >> 5;
    const int lane = threadIdx.x & 31;
    const int row  = blockIdx.x * 8 + wid;

    const float4* rp = (const float4*)(emb + (size_t)row * DIM);
    float4 x0 = rp[lane * 2 + 0];
    float4 x1 = rp[lane * 2 + 1];

    float ss = x0.x*x0.x + x0.y*x0.y + x0.z*x0.z + x0.w*x0.w
             + x1.x*x1.x + x1.y*x1.y + x1.z*x1.z + x1.w*x1.w;
#pragma unroll
    for (int off = 16; off >= 1; off >>= 1) ss += __shfl_xor(ss, off, 32);

    const float inv = 1.0f / fmaxf(sqrtf(ss), 1e-12f);
    float n[8] = { x0.x*inv, x0.y*inv, x0.z*inv, x0.w*inv,
                   x1.x*inv, x1.y*inv, x1.z*inv, x1.w*inv };

    v8h h;
#pragma unroll
    for (int i = 0; i < 8; i++) h[i] = (_Float16)n[i];
    *(v8h*)(eh + (size_t)row * DIM + lane * 8) = h;

    const int lbl = labels[row];
    float* ps = protoSum + (size_t)lbl * DIM + lane * 8;
#pragma unroll
    for (int i = 0; i < 8; i++) atomicAdd(ps + i, n[i]);
    if (lane == 0) atomicAdd(counts + lbl, 1.0f);
}

// ---- Kernel B: prototypes = normalize(sum / count), stored row-major f16 ---
__global__ __launch_bounds__(256)
void make_protos(const float* __restrict__ protoSum,
                 const float* __restrict__ counts,
                 _Float16*    __restrict__ ph)
{
    const int wid  = threadIdx.x >> 5;
    const int lane = threadIdx.x & 31;
    const int cls  = blockIdx.x * 8 + wid;

    const float cnt  = counts[cls];
    const float cinv = cnt > 0.0f ? 1.0f / cnt : 0.0f;

    const float4* rp = (const float4*)(protoSum + (size_t)cls * DIM);
    float4 x0 = rp[lane * 2 + 0];
    float4 x1 = rp[lane * 2 + 1];
    float m[8] = { x0.x*cinv, x0.y*cinv, x0.z*cinv, x0.w*cinv,
                   x1.x*cinv, x1.y*cinv, x1.z*cinv, x1.w*cinv };

    float ss = 0.0f;
#pragma unroll
    for (int i = 0; i < 8; i++) ss += m[i] * m[i];
#pragma unroll
    for (int off = 16; off >= 1; off >>= 1) ss += __shfl_xor(ss, off, 32);

    const float inv = 1.0f / fmaxf(sqrtf(ss), 1e-12f);
    v8h h;
#pragma unroll
    for (int i = 0; i < 8; i++) h[i] = (_Float16)(m[i] * inv);
    *(v8h*)(ph + (size_t)cls * DIM + lane * 8) = h;
}

#if HAVE_TDM
// TDM descriptor: 1D copy of CHUNK_ROWS*512B from global to LDS, with LDS
// padding of 16B after every 512B (bank-conflict-free 528B row stride).
__device__ __forceinline__ void tdm_load_chunk(const _Float16* gsrc, unsigned lds_off)
{
    const unsigned long long ga = (unsigned long long)(uintptr_t)gsrc;
    v4u g0;
    g0[0] = 1u;                                     // count=1, user descriptor
    g0[1] = lds_off;                                // LDS byte address
    g0[2] = (unsigned)(ga & 0xFFFFFFFFu);
    g0[3] = (unsigned)((ga >> 32) & 0x01FFFFFFu) | (2u << 30);  // type=2 (image)

    v8i g1;
    // data_size=3 (8B), pad_enable, pad_interval=6 (128 dw = 512B), pad_amount=3 (4 dw = 16B)
    g1[0] = (3 << 16) | (1 << 20) | (6 << 22) | (3 << 25);
    const unsigned td0 = (unsigned)(NCLS * DIM * 2 / 8);          // 65536 units of 8B
    const unsigned tl0 = (unsigned)(CHUNK_ROWS * DIM * 2 / 8);    // 2048 units of 8B
    g1[1] = (int)((td0 & 0xFFFFu) << 16);                         // tensor_dim0 lo16
    g1[2] = (int)((td0 >> 16) | (1u << 16));                      // tensor_dim0 hi16 | tensor_dim1=1
    g1[3] = (int)(tl0 << 16);                                     // tile_dim0
    g1[4] = 0;                                                    // tile_dim1/2 = 0
    g1[5] = (int)td0;                                             // tensor_dim0_stride lo32
    g1[6] = 0;
    g1[7] = 0;

    v4i z4 = {0, 0, 0, 0};
#if __clang_major__ >= 23
    v8i z8 = {0, 0, 0, 0, 0, 0, 0, 0};
    __builtin_amdgcn_tensor_load_to_lds(g0, g1, z4, z4, z8, 0);
#else
    __builtin_amdgcn_tensor_load_to_lds(g0, g1, z4, z4, 0);
#endif
}
__device__ __forceinline__ void tdm_wait()
{
#if __has_builtin(__builtin_amdgcn_s_wait_tensorcnt)
    __builtin_amdgcn_s_wait_tensorcnt(0);
#else
    asm volatile("s_wait_tensorcnt 0x0" ::: "memory");
#endif
}
#else
// Fallback: cooperative synchronous copy with the same padded layout.
__device__ __forceinline__ void copy_chunk(_Float16* sdst, const _Float16* gsrc, int tid)
{
    const int r = tid >> 3, p = tid & 7;          // 8 threads per 512B row
    const v8h* s = (const v8h*)(gsrc + r * DIM + p * 32);
    v8h*       d = (v8h*)(sdst + r * LROW + p * 32);
#pragma unroll
    for (int i = 0; i < 4; i++) d[i] = s[i];
}
#endif

// ---- Kernel C: fused cosine-GEMM (WMMA f16->f32, LDS-staged B via TDM) +
//      bounded-logit softmax accumulation + exact margin fix-up -------------
__global__ __launch_bounds__(256)
void arc_loss(const _Float16* __restrict__ eh,
              const _Float16* __restrict__ ph,
              const int*      __restrict__ labels,
              float*          __restrict__ nll)
{
    __shared__ __align__(128) _Float16 smem[2][CHUNK_ROWS * LROW];

    const int tid   = threadIdx.x;
    const int wid   = tid >> 5;
    const int lane  = tid & 31;
    const int rowb  = blockIdx.x * 256 + wid * 32;  // 32 rows per wave
    const int mrow  = lane & 15;
    const int khalf = lane >> 4;
    const int ncol  = lane & 15;
    const int ksel  = khalf * 8;

    // Hoist A: 2 row-tiles x (16 rows x 256 K) f16 in WMMA layout
    v16h a[2][8];
#pragma unroll
    for (int rt = 0; rt < 2; rt++) {
        const _Float16* arow = eh + (size_t)(rowb + rt * 16 + mrow) * DIM;
#pragma unroll
        for (int kc = 0; kc < 8; kc++) {
            v8h lo = *(const v8h*)(arow + kc * 32 + 8 * khalf);
            v8h hi = *(const v8h*)(arow + kc * 32 + 16 + 8 * khalf);
#pragma unroll
            for (int i = 0; i < 8; i++) { a[rt][kc][i] = lo[i]; a[rt][kc][8 + i] = hi[i]; }
        }
    }

    int   lbl[2][8];
    float S[2][8], ct[2][8];
#pragma unroll
    for (int rt = 0; rt < 2; rt++)
#pragma unroll
        for (int v = 0; v < 8; v++) {
            lbl[rt][v] = labels[rowb + rt * 16 + v + 8 * khalf];
            S[rt][v] = 0.0f; ct[rt][v] = 0.0f;
        }

#if HAVE_TDM
    if (wid == 0) tdm_load_chunk(ph, (unsigned)(uintptr_t)(void*)&smem[0][0]);
#endif

    for (int c = 0; c < NCHUNK; c++) {
#if HAVE_TDM
        if (wid == 0) tdm_wait();               // chunk c landed in LDS
        __syncthreads();                        // publish; prior compute done
        if (wid == 0 && c + 1 < NCHUNK)         // overlap next DMA with compute
            tdm_load_chunk(ph + (size_t)(c + 1) * CHUNK_ROWS * DIM,
                           (unsigned)(uintptr_t)(void*)&smem[(c + 1) & 1][0]);
        const _Float16* sb = &smem[c & 1][0];
#else
        __syncthreads();
        copy_chunk(&smem[0][0], ph + (size_t)c * CHUNK_ROWS * DIM, tid);
        __syncthreads();
        const _Float16* sb = &smem[0][0];
#endif

#pragma unroll
        for (int t = 0; t < CHUNK_ROWS / 16; t++) {
            const _Float16* bcol = sb + (t * 16 + ncol) * LROW + ksel;
            v8f acc0 = {}, acc1 = {};
#pragma unroll
            for (int kc = 0; kc < 8; kc++) {
                v8h blo = *(const v8h*)(bcol + kc * 32);
                v8h bhi = *(const v8h*)(bcol + kc * 32 + 16);
                v16h b;
#pragma unroll
                for (int i = 0; i < 8; i++) { b[i] = blo[i]; b[8 + i] = bhi[i]; }
                acc0 = __builtin_amdgcn_wmma_f32_16x16x32_f16(
                    false, a[0][kc], false, b, (short)0, acc0, false, false);
                acc1 = __builtin_amdgcn_wmma_f32_16x16x32_f16(
                    false, a[1][kc], false, b, (short)0, acc1, false, false);
            }
            const int n = c * CHUNK_ROWS + t * 16 + ncol;
#pragma unroll
            for (int v = 0; v < 8; v++) {
                const float c0 = acc0[v], c1 = acc1[v];
                S[0][v] += __expf(c0 * INV_T_);              // logits bounded: no max needed
                S[1][v] += __expf(c1 * INV_T_);
                ct[0][v] = (n == lbl[0][v]) ? c0 : ct[0][v]; // capture raw target cosine
                ct[1][v] = (n == lbl[1][v]) ? c1 : ct[1][v];
            }
        }
    }

    // Lane-half reductions, then margin math on 2 lanes only; exact fix-up of S
#pragma unroll
    for (int rt = 0; rt < 2; rt++)
#pragma unroll
        for (int v = 0; v < 8; v++) {
            float s = S[rt][v], t = ct[rt][v];
#pragma unroll
            for (int off = 8; off >= 1; off >>= 1) {
                s += __shfl_xor(s, off, 32);
                t += __shfl_xor(t, off, 32);
            }
            if ((lane & 15) == 0) {
                const float co  = t;
                const float sn  = sqrtf(fmaxf(0.0f, fminf(1.0f, 1.0f - co * co)));
                const float phi = co * COS_M_ - sn * SIN_M_;
                const float sel = (co - TH_ > 0.0f) ? phi : (co - MARGIN_);
                const float lt  = sel * INV_T_;
                const float Sf  = s - __expf(co * INV_T_) + __expf(lt);
                nll[rowb + rt * 16 + v + 8 * khalf] = __logf(Sf) - lt;
            }
        }
}

// ---- Kernel D: deterministic mean over 65536 NLL values ---------------------
__global__ __launch_bounds__(256)
void reduce_mean(const float* __restrict__ nll, float* __restrict__ out)
{
    __shared__ float sm[256];
    float s = 0.0f;
    for (int i = threadIdx.x; i < NROWS; i += 256) s += nll[i];
    sm[threadIdx.x] = s;
    __syncthreads();
    for (int st = 128; st > 0; st >>= 1) {
        if (threadIdx.x < st) sm[threadIdx.x] += sm[threadIdx.x + st];
        __syncthreads();
    }
    if (threadIdx.x == 0) out[0] = sm[0] / (float)NROWS;
}

extern "C" void kernel_launch(void* const* d_in, const int* in_sizes, int n_in,
                              void* d_out, int out_size, void* d_ws, size_t ws_size,
                              hipStream_t stream)
{
    const float* emb    = (const float*)d_in[0];
    const int*   labels = (const int*)d_in[1];

    char* ws = (char*)d_ws;
    // layout: eh 32MB | protoSum 1MB | counts 4KB | proto f16 0.5MB | nll 256KB
    _Float16* eh       = (_Float16*)ws;
    float*    protoSum = (float*)(ws + (size_t)32 * 1024 * 1024);
    float*    counts   = (float*)(ws + (size_t)32 * 1024 * 1024 + 1048576);
    _Float16* ph       = (_Float16*)(ws + (size_t)32 * 1024 * 1024 + 1048576 + 4096);
    float*    nll      = (float*)(ws + (size_t)32 * 1024 * 1024 + 1048576 + 4096 + 524288);

    const int nz = NCLS * DIM + NCLS;   // protoSum and counts are contiguous
    zero_f32<<<(nz + 255) / 256, 256, 0, stream>>>(protoSum, nz);
    normalize_scatter<<<NROWS / 8, 256, 0, stream>>>(emb, labels, eh, protoSum, counts);
    make_protos<<<NCLS / 8, 256, 0, stream>>>(protoSum, counts, ph);
    arc_loss<<<NROWS / 256, 256, 0, stream>>>(eh, ph, labels, nll);
    reduce_mean<<<1, 256, 0, stream>>>(nll, (float*)d_out);
}
